// CausalSelfAttention_5669356835512
// MI455X (gfx1250) — matrix operand order repaired and checked by reading, not compiled, not run on hardware
//
#include <hip/hip_runtime.h>

// ---------------------------------------------------------------------------
// CausalSelfAttention forward for MI455X (gfx1250, wave32, WMMA bf16)
// B=2, T=2048, C=1024, H=16, D=64
// GEMMs: 16x64 strip per wave (4 WMMA accumulators), ping-pong double-buffered
// (2x-unrolled pipeline) so loads overlap WMMA with zero rotation moves.
// ---------------------------------------------------------------------------

typedef __bf16 bf16_t;
typedef __attribute__((ext_vector_type(16))) __bf16 bf16x16;
typedef __attribute__((ext_vector_type(8)))  __bf16 bf16x8;
typedef __attribute__((ext_vector_type(8)))  float  f32x8;

#define NEG_INF (-__builtin_inff())

constexpr int Bn = 2, Tn = 2048, Cn = 1024, Hn = 16, Dn = 64;
constexpr int Rows = Bn * Tn;          // 4096 token rows
constexpr int C3 = 3 * Cn;             // 3072

// ---- WMMA wrapper: D = A(16x32 bf16) * B(32x16 bf16) + C(16x16 f32) -------
static __device__ __forceinline__ f32x8 wmma_bf16(bf16x16 a, bf16x16 b, f32x8 c) {
  return __builtin_amdgcn_wmma_f32_16x16x32_bf16(
      /*neg_a=*/false, a, /*neg_b=*/false, b,
      /*c_mod=*/(short)0, c, /*reuse_a=*/false, /*reuse_b=*/false);
}

// A fragment (16x32, row-major source). rowbase = &A[tile_row + (lane&15)][0].
// lane half hf = lane>>4.  ISA layout: v0..3 hold K = hf*8..hf*8+7,
// v4..7 hold K = 16+hf*8 .. 16+hf*8+7 (2 bf16 per VGPR).
static __device__ __forceinline__ bf16x16
load_a_frag(const bf16_t* __restrict__ rowbase, int k0, int hf) {
  bf16x8 lo = *(const bf16x8*)(rowbase + k0 + hf * 8);
  bf16x8 hi = *(const bf16x8*)(rowbase + k0 + 16 + hf * 8);
  bf16x16 r;
#pragma unroll
  for (int i = 0; i < 8; ++i) { r[i] = lo[i]; r[i + 8] = hi[i]; }
  return r;
}

// B fragment (32x16).  Source stored n-major: nrowbase = &Bt[n][0] where
// Bt[n][k] == B[k][n].  Same K layout as A: v0..3 K=hf*8.., v4..7 K=16+hf*8..
static __device__ __forceinline__ bf16x16
load_b_frag(const bf16_t* __restrict__ nrowbase, int k0, int hf) {
  const bf16_t* p = nrowbase + k0 + hf * 8;
  bf16x8 lo = *(const bf16x8*)(p);
  bf16x8 hi = *(const bf16x8*)(p + 16);
  bf16x16 r;
#pragma unroll
  for (int i = 0; i < 8; ++i) { r[i] = lo[i]; r[i + 8] = hi[i]; }
  return r;
}

// ---------------------------------------------------------------------------
// Prep kernels: fp32 -> bf16 convert / transpose-convert
// ---------------------------------------------------------------------------
__global__ void k_cvt(const float* __restrict__ in, bf16_t* __restrict__ out, int n) {
  for (int i = blockIdx.x * blockDim.x + threadIdx.x; i < n; i += gridDim.x * blockDim.x)
    out[i] = (bf16_t)in[i];
}

// in[k*N + n] (K x N, row major)  ->  out[n*K + k]  (N x K, row major)
__global__ void k_tr(const float* __restrict__ in, bf16_t* __restrict__ out, int K, int N) {
  int total = K * N;
  for (int i = blockIdx.x * blockDim.x + threadIdx.x; i < total; i += gridDim.x * blockDim.x) {
    int k = i / N, n = i - k * N;
    out[n * K + k] = (bf16_t)in[i];
  }
}

// ---------------------------------------------------------------------------
// Ping-pong pipelined 16x64-strip GEMM: acc[4] += A_strip(16xCn) x B(Cnx64)
// Buffer set 0 is consumed while set 1 loads, and vice versa; loads assign
// directly into the loop-carried fragments (no rotation moves).
// ---------------------------------------------------------------------------
static __device__ __forceinline__ void
gemm_strip(const bf16_t* __restrict__ arow, const bf16_t* __restrict__ brow,
           int hf, f32x8 acc[4]) {
  const bf16_t* br0 = brow;
  const bf16_t* br1 = brow + 16 * Cn;
  const bf16_t* br2 = brow + 32 * Cn;
  const bf16_t* br3 = brow + 48 * Cn;

  bf16x16 a0  = load_a_frag(arow, 0, hf);
  bf16x16 b00 = load_b_frag(br0, 0, hf);
  bf16x16 b01 = load_b_frag(br1, 0, hf);
  bf16x16 b02 = load_b_frag(br2, 0, hf);
  bf16x16 b03 = load_b_frag(br3, 0, hf);

#pragma unroll 1
  for (int k0 = 0; k0 < Cn - 64; k0 += 64) {
    __builtin_prefetch(arow + k0 + 192, 0, 3);   // WGP-level prefetch
    __builtin_prefetch(br0 + k0 + 192, 0, 3);
    // load set 1 (k0+32) while consuming set 0 (k0)
    bf16x16 a1  = load_a_frag(arow, k0 + 32, hf);
    bf16x16 b10 = load_b_frag(br0, k0 + 32, hf);
    bf16x16 b11 = load_b_frag(br1, k0 + 32, hf);
    bf16x16 b12 = load_b_frag(br2, k0 + 32, hf);
    bf16x16 b13 = load_b_frag(br3, k0 + 32, hf);
    acc[0] = wmma_bf16(a0, b00, acc[0]);
    acc[1] = wmma_bf16(a0, b01, acc[1]);
    acc[2] = wmma_bf16(a0, b02, acc[2]);
    acc[3] = wmma_bf16(a0, b03, acc[3]);
    // load set 0 (k0+64) while consuming set 1 (k0+32)
    a0  = load_a_frag(arow, k0 + 64, hf);
    b00 = load_b_frag(br0, k0 + 64, hf);
    b01 = load_b_frag(br1, k0 + 64, hf);
    b02 = load_b_frag(br2, k0 + 64, hf);
    b03 = load_b_frag(br3, k0 + 64, hf);
    acc[0] = wmma_bf16(a1, b10, acc[0]);
    acc[1] = wmma_bf16(a1, b11, acc[1]);
    acc[2] = wmma_bf16(a1, b12, acc[2]);
    acc[3] = wmma_bf16(a1, b13, acc[3]);
  }
  // tail: last 64 of K (set 0 holds Cn-64; load Cn-32 as set 1)
  {
    bf16x16 a1  = load_a_frag(arow, Cn - 32, hf);
    bf16x16 b10 = load_b_frag(br0, Cn - 32, hf);
    bf16x16 b11 = load_b_frag(br1, Cn - 32, hf);
    bf16x16 b12 = load_b_frag(br2, Cn - 32, hf);
    bf16x16 b13 = load_b_frag(br3, Cn - 32, hf);
    acc[0] = wmma_bf16(a0, b00, acc[0]);
    acc[1] = wmma_bf16(a0, b01, acc[1]);
    acc[2] = wmma_bf16(a0, b02, acc[2]);
    acc[3] = wmma_bf16(a0, b03, acc[3]);
    acc[0] = wmma_bf16(a1, b10, acc[0]);
    acc[1] = wmma_bf16(a1, b11, acc[1]);
    acc[2] = wmma_bf16(a1, b12, acc[2]);
    acc[3] = wmma_bf16(a1, b13, acc[3]);
  }
}

// ---------------------------------------------------------------------------
// QKV GEMM:  qkv[4096,3072] = xb[4096,1024] @ W_attn + b_attn
// One wave computes a 16x64 strip.  A 64-col strip lies in exactly one of the
// Q/K/V segments and one head, so the scatter is wave-uniform.
// Q,K -> [B,H,T,D] bf16 ; V -> [B,H,D,T] bf16 (transposed)
// ---------------------------------------------------------------------------
__global__ __launch_bounds__(128) void k_qkv(const bf16_t* __restrict__ xb,
                                             const bf16_t* __restrict__ Wt,   // [3072,1024]
                                             const float*  __restrict__ bias, // [3072]
                                             bf16_t* __restrict__ Q,
                                             bf16_t* __restrict__ K,
                                             bf16_t* __restrict__ V) {
  const int lane = threadIdx.x & 31;
  const int wave = threadIdx.x >> 5;
  const int wtile = blockIdx.x * 4 + wave;   // 0 .. 256*48-1
  const int NGN = C3 / 64;                   // 48 strips along N
  const int mt = wtile / NGN, ng = wtile % NGN;
  const int ln = lane & 15, hf = lane >> 4;

  const bf16_t* arow = xb + (mt * 16 + ln) * Cn;
  const bf16_t* brow = Wt + (ng * 64 + ln) * Cn;   // +j*16*Cn for sub-tile j

  f32x8 acc[4] = {{}, {}, {}, {}};
  gemm_strip(arow, brow, hf, acc);

  // epilogue: strip covers cols [ng*64, ng*64+64) -> one segment, one head
  const int col0 = ng * 64;
  const int seg  = col0 >> 10;               // 0=Q 1=K 2=V (wave-uniform)
  const int h    = (col0 & 1023) >> 6;       // head (wave-uniform)
  if (seg == 0) {
#pragma unroll
    for (int j = 0; j < 4; ++j) {
      const float bv = bias[col0 + j * 16 + ln];
      const int d = j * 16 + ln;
#pragma unroll
      for (int r = 0; r < 8; ++r) {
        int row = mt * 16 + r + hf * 8;
        int b = row >> 11, t = row & 2047;
        Q[(((b * Hn + h) * Tn) + t) * Dn + d] = (bf16_t)(acc[j][r] + bv);
      }
    }
  } else if (seg == 1) {
#pragma unroll
    for (int j = 0; j < 4; ++j) {
      const float bv = bias[col0 + j * 16 + ln];
      const int d = j * 16 + ln;
#pragma unroll
      for (int r = 0; r < 8; ++r) {
        int row = mt * 16 + r + hf * 8;
        int b = row >> 11, t = row & 2047;
        K[(((b * Hn + h) * Tn) + t) * Dn + d] = (bf16_t)(acc[j][r] + bv);
      }
    }
  } else {
#pragma unroll
    for (int j = 0; j < 4; ++j) {
      const float bv = bias[col0 + j * 16 + ln];
      const int d = j * 16 + ln;
#pragma unroll
      for (int r = 0; r < 8; ++r) {
        int row = mt * 16 + r + hf * 8;
        int b = row >> 11, t = row & 2047;
        V[(((b * Hn + h) * Dn) + d) * Tn + t] = (bf16_t)(acc[j][r] + bv);
      }
    }
  }
}

// ---------------------------------------------------------------------------
// Attention: per block = one (b,h,qtile).  4 waves, full 16x2048 score row in
// LDS (128KB, CDNA5 WGP has 320KB), exact softmax, then P@V with WMMA.
// yb output: [4096, 1024] bf16 token-major (A matrix of the proj GEMM).
// ---------------------------------------------------------------------------
__global__ __launch_bounds__(128) void k_attn(const bf16_t* __restrict__ Q,   // [B,H,T,D]
                                              const bf16_t* __restrict__ K,   // [B,H,T,D]
                                              const bf16_t* __restrict__ Vt,  // [B,H,D,T]
                                              bf16_t* __restrict__ yb) {      // [Rows, C]
  extern __shared__ float smem[];
  float* S      = smem;                    // 16 x 2048 scores / probs
  float* redm   = smem + 16 * Tn;          // 128 partial maxima
  float* reds   = redm + 128;              // 128 partial sums
  float* rowinv = reds + 128;              // 16 inverse row sums

  const int bh = blockIdx.x >> 7;          // 0..31 = b*16+h
  const int qt = blockIdx.x & 127;         // query tile
  const int lane = threadIdx.x & 31;
  const int wave = threadIdx.x >> 5;
  const int ln = lane & 15, hf = lane >> 4;
  const int qmax = qt * 16 + 15;
  const float scale = 0.125f;              // 1/sqrt(64)

  const bf16_t* Qh = Q  + bh * Tn * Dn;
  const bf16_t* Kh = K  + bh * Tn * Dn;
  const bf16_t* Vh = Vt + bh * Dn * Tn;

  // ---- phase 1: S = (q @ k^T) * scale, causal-masked, into LDS -----------
  const bf16_t* qrow = Qh + (qt * 16 + ln) * Dn;
  bf16x16 a0 = load_a_frag(qrow, 0, hf);   // D chunk 0..31
  bf16x16 a1 = load_a_frag(qrow, 32, hf);  // D chunk 32..63

  for (int kt = wave * 32; kt < wave * 32 + 32; ++kt) {   // 32 key tiles/wave
    const int kb = kt * 16;
    if (kb > qmax) {                       // wave-uniform: fully masked tile
#pragma unroll
      for (int r = 0; r < 8; ++r) S[(r + hf * 8) * Tn + kb + ln] = NEG_INF;
      continue;
    }
    const bf16_t* krow = Kh + (kb + ln) * Dn;   // B fragment source
    f32x8 acc = {};
    acc = wmma_bf16(a0, load_b_frag(krow, 0, hf), acc);
    acc = wmma_bf16(a1, load_b_frag(krow, 32, hf), acc);
#pragma unroll
    for (int r = 0; r < 8; ++r) {
      int M = r + hf * 8;
      int kj = kb + ln;
      S[M * Tn + kj] = (kj <= qt * 16 + M) ? acc[r] * scale : NEG_INF;
    }
  }
  __syncthreads();

  // ---- phase 2: exact softmax over each of the 16 rows -------------------
  {
    const int row = threadIdx.x >> 3;      // 16 rows
    const int seg = threadIdx.x & 7;       // 8 segments x 256 cols
    float* Sr = S + row * Tn;
    float mx = NEG_INF;
    for (int j = seg * 256; j < seg * 256 + 256; ++j) mx = fmaxf(mx, Sr[j]);
    redm[row * 8 + seg] = mx;
    __syncthreads();
    float rm = redm[row * 8];
#pragma unroll
    for (int i = 1; i < 8; ++i) rm = fmaxf(rm, redm[row * 8 + i]);
    float sum = 0.f;
    for (int j = seg * 256; j < seg * 256 + 256; ++j) {
      float e = __expf(Sr[j] - rm);
      Sr[j] = e;                           // overwrite score with prob
      sum += e;
    }
    reds[row * 8 + seg] = sum;
    __syncthreads();
    if (threadIdx.x < 16) {
      float s = 0.f;
#pragma unroll
      for (int i = 0; i < 8; ++i) s += reds[threadIdx.x * 8 + i];
      rowinv[threadIdx.x] = 1.0f / s;
    }
  }
  __syncthreads();

  // ---- phase 3: y(16x64) = P(16x2048) @ V(2048x64); wave w -> d-tile w ----
  f32x8 acc = {};
  const bf16_t* vrow = Vh + (wave * 16 + ln) * Tn;  // B fragment: row d of V^T
  for (int kc = 0; kc < Tn / 32; ++kc) {
    const int kb = kc * 32;
    if (kb > qmax) break;                  // block-uniform causal skip
    const float* Pr = S + ln * Tn + kb;    // A fragment: probs (fp32->bf16)
    f32x8 lo = *(const f32x8*)(Pr + hf * 8);
    f32x8 hi = *(const f32x8*)(Pr + 16 + hf * 8);
    bf16x16 a;
#pragma unroll
    for (int i = 0; i < 8; ++i) { a[i] = (bf16_t)lo[i]; a[i + 8] = (bf16_t)hi[i]; }
    acc = wmma_bf16(a, load_b_frag(vrow, kb, hf), acc);
  }
  // epilogue: y[b, q, h*64 + d] with softmax normalization
  const int hq = bh & 15, bq = bh >> 4;
#pragma unroll
  for (int r = 0; r < 8; ++r) {
    int M = r + hf * 8;
    int q = qt * 16 + M;
    float v = acc[r] * rowinv[M];
    yb[(bq * Tn + q) * Cn + hq * 64 + wave * 16 + ln] = (bf16_t)v;
  }
}

// ---------------------------------------------------------------------------
// Output projection: out[4096,1024] = yb @ W_proj + b_proj  (fp32 out)
// One wave computes a 16x64 strip (pipelined core).
// ---------------------------------------------------------------------------
__global__ __launch_bounds__(128) void k_proj(const bf16_t* __restrict__ yb,
                                              const bf16_t* __restrict__ Wpt,  // [1024,1024] transposed
                                              const float*  __restrict__ bias,
                                              float* __restrict__ out) {
  const int lane = threadIdx.x & 31;
  const int wave = threadIdx.x >> 5;
  const int wtile = blockIdx.x * 4 + wave;   // 0 .. 256*16-1
  const int NGN = Cn / 64;                   // 16 strips along N
  const int mt = wtile / NGN, ng = wtile % NGN;
  const int ln = lane & 15, hf = lane >> 4;

  const bf16_t* arow = yb  + (mt * 16 + ln) * Cn;
  const bf16_t* brow = Wpt + (ng * 64 + ln) * Cn;

  f32x8 acc[4] = {{}, {}, {}, {}};
  gemm_strip(arow, brow, hf, acc);

#pragma unroll
  for (int j = 0; j < 4; ++j) {
    const int col = ng * 64 + j * 16 + ln;
    const float bv = bias[col];
#pragma unroll
    for (int r = 0; r < 8; ++r) {
      int row = mt * 16 + r + hf * 8;
      out[row * Cn + col] = acc[j][r] + bv;
    }
  }
}

// ---------------------------------------------------------------------------
// Host launcher
// ---------------------------------------------------------------------------
extern "C" void kernel_launch(void* const* d_in, const int* in_sizes, int n_in,
                              void* d_out, int out_size, void* d_ws, size_t ws_size,
                              hipStream_t stream) {
  const float* x      = (const float*)d_in[0];   // [B,T,C]
  const float* W_attn = (const float*)d_in[1];   // [C,3C]
  const float* b_attn = (const float*)d_in[2];   // [3C]
  const float* W_proj = (const float*)d_in[3];   // [C,C]
  const float* b_proj = (const float*)d_in[4];   // [C]
  float* out = (float*)d_out;

  char* ws = (char*)d_ws;
  bf16_t* xb  = (bf16_t*)(ws);                        //  8 MB  [4096,1024]
  bf16_t* Wat = (bf16_t*)(ws + (size_t)8  * 1048576); //  6 MB  [3072,1024]
  bf16_t* Wpt = (bf16_t*)(ws + (size_t)14 * 1048576); //  2 MB  [1024,1024]
  bf16_t* Qb  = (bf16_t*)(ws + (size_t)16 * 1048576); //  8 MB  [B,H,T,D]
  bf16_t* Kb  = (bf16_t*)(ws + (size_t)24 * 1048576); //  8 MB  [B,H,T,D]
  bf16_t* Vb  = (bf16_t*)(ws + (size_t)32 * 1048576); //  8 MB  [B,H,D,T]
  bf16_t* yb  = (bf16_t*)(ws + (size_t)40 * 1048576); //  8 MB  [4096,1024]

  // prep: convert / transpose to bf16
  k_cvt<<<1024, 256, 0, stream>>>(x, xb, Rows * Cn);
  k_tr <<<2048, 256, 0, stream>>>(W_attn, Wat, Cn, C3);
  k_tr <<<1024, 256, 0, stream>>>(W_proj, Wpt, Cn, Cn);

  // QKV GEMM: 256 M-tiles x 48 N-strips = 12288 wave-strips, 4 waves/block
  k_qkv<<<12288 / 4, 128, 0, stream>>>(xb, Wat, b_attn, Qb, Kb, Vb);

  // attention: B*H*(T/16) = 4096 blocks, 128 threads, ~129 KB dynamic LDS
  size_t smem_bytes = (size_t)(16 * Tn + 128 + 128 + 16) * sizeof(float);
  k_attn<<<Bn * Hn * (Tn / 16), 128, smem_bytes, stream>>>(Qb, Kb, Vb, yb);

  // projection: 256 M-tiles x 16 N-strips = 4096 wave-strips, 4 waves/block
  k_proj<<<4096 / 4, 128, 0, stream>>>(yb, Wpt, b_proj, out);
}
